// LogicGatedSpikingSelfAttention_1176821039346
// MI455X (gfx1250) — 16-bit and 8-bit matrix operand order repaired and checked by reading, compiled, not run on hardware
//
#include <hip/hip_runtime.h>
#include <stdint.h>


typedef __attribute__((ext_vector_type(16))) __bf16 v16bf;
typedef __attribute__((ext_vector_type(8)))  float  v8f;
typedef __attribute__((ext_vector_type(8)))  int    v8i;

#define N_TOK   1024
#define DMODEL  1024
#define NHEAD   16
#define HD      64
#define NBATCH  4
#define ROWS    4096
#define SCALE_ATTN 0.5946035575013605f

__device__ __forceinline__ v16bf mk16bf(uint4 a, uint4 b) {
  union { uint4 u[2]; v16bf v; } t; t.u[0] = a; t.u[1] = b; return t.v;
}
__device__ __forceinline__ v8i mk8i(uint4 a, uint4 b) {
  union { uint4 u[2]; v8i v; } t; t.u[0] = a; t.u[1] = b; return t.v;
}
__device__ __forceinline__ v8i mk8i2(uint2 a, uint2 b, uint2 c, uint2 d) {
  union { uint2 u[4]; v8i v; } t; t.u[0]=a; t.u[1]=b; t.u[2]=c; t.u[3]=d; return t.v;
}

__device__ __forceinline__ void async_copy_b128(unsigned lds_off, const void* gaddr) {
  asm volatile("global_load_async_to_lds_b128 %0, %1, off"
               :: "v"(lds_off), "v"(gaddr) : "memory");
}
__device__ __forceinline__ void wait_asynccnt0() {
  asm volatile("s_wait_asynccnt 0" ::: "memory");
}

__global__ __launch_bounds__(256) void f32_to_bf16_kernel(
    const float* __restrict__ src, uint16_t* __restrict__ dst, int n) {
  int i = blockIdx.x * 256 + threadIdx.x;
  if (i < n) {
    uint32_t u = __float_as_uint(src[i]);
    uint32_t r = (u + 0x7FFFu + ((u >> 16) & 1u)) >> 16;
    dst[i] = (uint16_t)r;
  }
}

__global__ __launch_bounds__(256) void gemm_bf16_kernel(
    const uint16_t* __restrict__ A, const uint16_t* __restrict__ W,
    const float* __restrict__ bias, float* __restrict__ C) {
  __shared__ __align__(16) uint16_t Blds[128 * 64];
  const int tid  = threadIdx.x;
  const int wave = tid >> 5;
  const int lane = tid & 31;
  const int l16  = lane & 15;
  const int hi   = lane >> 4;
  const int rowBase = blockIdx.x * 128 + wave * 16;
  const int colBase = blockIdx.y * 128;

  const uint16_t* arow = A + (size_t)(rowBase + l16) * DMODEL;
  const int aseg = hi * 8;
  const unsigned ldsBase = (unsigned)(uintptr_t)(&Blds[0]);

  v8f acc[8];
#pragma unroll
  for (int t = 0; t < 8; ++t) acc[t] = (v8f){0.f,0.f,0.f,0.f,0.f,0.f,0.f,0.f};

  for (int k0 = 0; k0 < DMODEL; k0 += 64) {
#pragma unroll
    for (int s = 0; s < 4; ++s) {
      int seg  = tid * 4 + s;
      int c    = seg >> 3;
      int part = seg & 7;
      const uint16_t* gsrc = W + (size_t)(colBase + c) * DMODEL + k0 + part * 8;
      async_copy_b128(ldsBase + (unsigned)(c * 64 + part * 8) * 2, gsrc);
    }
    __builtin_prefetch(arow + k0 + 64, 0, 1);
    wait_asynccnt0();
    __syncthreads();

#pragma unroll
    for (int ks = 0; ks < 64; ks += 32) {
      uint4 a0 = *(const uint4*)(arow + k0 + ks + aseg);
      uint4 a1 = *(const uint4*)(arow + k0 + ks + aseg + 16);
      v16bf af = mk16bf(a0, a1);
#pragma unroll
      for (int t = 0; t < 8; ++t) {
        const uint16_t* bp = &Blds[(t * 16 + l16) * 64 + ks + hi * 8];
        uint4 b0 = *(const uint4*)(bp);
        uint4 b1 = *(const uint4*)(bp + 16);
        v16bf bf = mk16bf(b0, b1);
        acc[t] = __builtin_amdgcn_wmma_f32_16x16x32_bf16(
            false, af, false, bf, (short)0, acc[t], false, false);
      }
    }
    __syncthreads();
  }
#pragma unroll
  for (int t = 0; t < 8; ++t) {
    int col = colBase + t * 16 + l16;
    float bv = bias[col];
#pragma unroll
    for (int i = 0; i < 8; ++i) {
      int row = rowBase + i + hi * 8;
      C[(size_t)row * DMODEL + col] = acc[t][i] + bv;
    }
  }
}

__global__ __launch_bounds__(256) void bn_stats_kernel(
    const float* __restrict__ Y, const float* __restrict__ gamma,
    const float* __restrict__ beta, float* __restrict__ colA,
    float* __restrict__ colB) {
  int c = blockIdx.x * 256 + threadIdx.x;
  float s = 0.f, ss = 0.f;
  for (int r = 0; r < ROWS; ++r) {
    float y = Y[(size_t)r * DMODEL + c];
    s += y; ss += y * y;
  }
  float m = s * (1.0f / ROWS);
  float var = ss * (1.0f / ROWS) - m * m;
  float a = gamma[c] * rsqrtf(var + 1e-5f);
  colA[c] = a;
  colB[c] = beta[c] - m * a;
}

__global__ __launch_bounds__(256) void binarize_kernel(
    const float* __restrict__ Y, const float* __restrict__ colA,
    const float* __restrict__ colB, uint8_t* __restrict__ dstU8,
    float* __restrict__ dstF32, int mode) {
  int idx = blockIdx.x * 256 + threadIdx.x;
  int row = idx >> 10;
  int col = idx & 1023;
  float v = (Y[idx] * colA[col] + colB[col]) * 0.5f;
  int bit = (v >= 1.0f) ? 1 : 0;
  if (mode == 2) { dstF32[idx] = (float)bit; return; }
  int b = row >> 10, n = row & 1023;
  int h = col >> 6,  d = col & 63;
  int bh = b * NHEAD + h;
  size_t o = (mode == 1) ? (((size_t)bh * HD + d) * N_TOK + n)
                         : (((size_t)bh * N_TOK + n) * HD + d);
  dstU8[o] = (uint8_t)bit;
}

__global__ __launch_bounds__(256) void energy_kernel(
    const uint8_t* __restrict__ q, const uint8_t* __restrict__ k,
    float* __restrict__ energy) {
  int bh = blockIdx.x;
  const unsigned* qu = (const unsigned*)(q + (size_t)bh * N_TOK * HD);
  const unsigned* ku = (const unsigned*)(k + (size_t)bh * N_TOK * HD);
  int acc = 0;
  for (int i = threadIdx.x; i < (N_TOK * HD) / 4; i += 256)
    acc += __popc(qu[i] & ku[i]);
  __shared__ int red[256];
  red[threadIdx.x] = acc;
  __syncthreads();
  for (int s = 128; s > 0; s >>= 1) {
    if (threadIdx.x < s) red[threadIdx.x] += red[threadIdx.x + s];
    __syncthreads();
  }
  if (threadIdx.x == 0) energy[bh] = (float)red[0] * (1.0f / N_TOK);
}

__global__ __launch_bounds__(256) void gate_kernel(
    const float* __restrict__ Wg, const float* __restrict__ bg,
    const float* __restrict__ energy, float* __restrict__ gate) {
  __shared__ float wred[16][16];
  int g = threadIdx.x >> 4, h = threadIdx.x & 15;
  float s = 0.f;
  for (int r = 0; r < 64; ++r) s += Wg[(size_t)g * DMODEL + r * 16 + h];
  wred[g][h] = s;
  __syncthreads();
  if (threadIdx.x < 64) {
    int b = threadIdx.x >> 4, gg = threadIdx.x & 15;
    float acc = bg[gg];
    for (int hh = 0; hh < 16; ++hh) acc += wred[gg][hh] * energy[b * 16 + hh];
    gate[threadIdx.x] = (acc >= 0.5f) ? 1.0f : 0.0f;
  }
}

__global__ __launch_bounds__(256) void attention_kernel(
    const uint8_t* __restrict__ qbin, const uint8_t* __restrict__ kbin,
    const uint8_t* __restrict__ vT, const float* __restrict__ gate,
    uint16_t* __restrict__ s2) {
  __shared__ uint8_t slds[8][16 * 64];
  const int bh   = blockIdx.x;
  const int wave = threadIdx.x >> 5;
  const int lane = threadIdx.x & 31;
  const int l16  = lane & 15;
  const int hi   = lane >> 4;
  const int b = bh >> 4, h = bh & 15;
  const int nBase = blockIdx.y * 128 + wave * 16;
  const float g = gate[bh];
  const int s8 = hi * 8;

  const uint8_t* qp = qbin + ((size_t)bh * N_TOK + nBase + l16) * HD;
  v8i af = mk8i2(*(const uint2*)(qp + s8),
                 *(const uint2*)(qp + s8 + 16),
                 *(const uint2*)(qp + s8 + 32),
                 *(const uint2*)(qp + s8 + 48));

  v8i xacc[4];
#pragma unroll
  for (int t = 0; t < 4; ++t) xacc[t] = (v8i){0,0,0,0,0,0,0,0};

  if (g != 0.0f) {
    uint8_t* sl = &slds[wave][0];
    for (int m0 = 0; m0 < N_TOK; m0 += 64) {
      v8i sacc[4];
#pragma unroll
      for (int t = 0; t < 4; ++t) {
        const uint8_t* kp = kbin + ((size_t)bh * N_TOK + m0 + t * 16 + l16) * HD + s8;
        v8i bf = mk8i2(*(const uint2*)kp, *(const uint2*)(kp + 16), *(const uint2*)(kp + 32), *(const uint2*)(kp + 48));
        v8i z = (v8i){0,0,0,0,0,0,0,0};
        sacc[t] = __builtin_amdgcn_wmma_i32_16x16x64_iu8(
            false, af, false, bf, z, false, false);
      }
#pragma unroll
      for (int t = 0; t < 4; ++t)
#pragma unroll
        for (int i = 0; i < 8; ++i)
          sl[(i + hi * 8) * 64 + t * 16 + l16] = (uint8_t)sacc[t][i];
      const uint8_t* sr = sl + l16 * 64 + s8;
      v8i sf = mk8i2(*(const uint2*)(sr),
                     *(const uint2*)(sr + 16),
                     *(const uint2*)(sr + 32),
                     *(const uint2*)(sr + 48));
#pragma unroll
      for (int t = 0; t < 4; ++t) {
        const uint8_t* vp = vT + ((size_t)bh * HD + t * 16 + l16) * N_TOK + m0 + s8;
        v8i bf = mk8i2(*(const uint2*)vp, *(const uint2*)(vp + 16), *(const uint2*)(vp + 32), *(const uint2*)(vp + 48));
        xacc[t] = __builtin_amdgcn_wmma_i32_16x16x64_iu8(
            false, sf, false, bf, xacc[t], false, false);
      }
    }
  }

  const float sg = SCALE_ATTN * g;
  const int rowG = b * N_TOK + nBase;
#pragma unroll
  for (int t = 0; t < 4; ++t) {
    int col = h * HD + t * 16 + l16;
#pragma unroll
    for (int i = 0; i < 8; ++i) {
      int row = rowG + i + hi * 8;
      float xv = sg * (float)xacc[t][i];
      s2[(size_t)row * DMODEL + col] = (xv >= 1.0f) ? (uint16_t)0x3F80 : (uint16_t)0;
    }
  }
}

extern "C" void kernel_launch(void* const* d_in, const int* in_sizes, int n_in,
                              void* d_out, int out_size, void* d_ws, size_t ws_size,
                              hipStream_t stream) {
  const float* x     = (const float*)d_in[0];
  const float* Wq    = (const float*)d_in[1];
  const float* bq    = (const float*)d_in[2];
  const float* gq    = (const float*)d_in[3];
  const float* betaq = (const float*)d_in[4];
  const float* Wk    = (const float*)d_in[5];
  const float* bk    = (const float*)d_in[6];
  const float* gk    = (const float*)d_in[7];
  const float* betak = (const float*)d_in[8];
  const float* Wv    = (const float*)d_in[9];
  const float* bv    = (const float*)d_in[10];
  const float* gv    = (const float*)d_in[11];
  const float* betav = (const float*)d_in[12];
  const float* Wp    = (const float*)d_in[13];
  const float* bp    = (const float*)d_in[14];
  const float* gp    = (const float*)d_in[15];
  const float* betap = (const float*)d_in[16];
  const float* Wg    = (const float*)d_in[17];
  const float* bg    = (const float*)d_in[18];

  char* ws = (char*)d_ws;
  size_t off = 0;
  auto alloc = [&](size_t bytes) -> void* {
    void* p = ws + off;
    off = (off + bytes + 255) & ~(size_t)255;
    return p;
  };
  uint16_t* xb   = (uint16_t*)alloc((size_t)ROWS * DMODEL * 2);
  uint16_t* wb[4];
  for (int i = 0; i < 4; ++i) wb[i] = (uint16_t*)alloc((size_t)DMODEL * DMODEL * 2);
  float*    Y    = (float*)alloc((size_t)ROWS * DMODEL * 4);
  uint8_t*  qbin = (uint8_t*)alloc((size_t)ROWS * DMODEL);
  uint8_t*  kbin = (uint8_t*)alloc((size_t)ROWS * DMODEL);
  uint8_t*  vTb  = (uint8_t*)alloc((size_t)ROWS * DMODEL);
  uint16_t* s2   = (uint16_t*)alloc((size_t)ROWS * DMODEL * 2);
  float*    colA = (float*)alloc(DMODEL * 4);
  float*    colB = (float*)alloc(DMODEL * 4);
  float*    ener = (float*)alloc(256);
  float*    gat  = (float*)alloc(256);

  const dim3 gemmGrid(ROWS / 128, DMODEL / 128);

  f32_to_bf16_kernel<<<(ROWS * DMODEL + 255) / 256, 256, 0, stream>>>(x, xb, ROWS * DMODEL);
  const float* Ws[4] = {Wq, Wk, Wv, Wp};
  for (int i = 0; i < 4; ++i)
    f32_to_bf16_kernel<<<(DMODEL * DMODEL + 255) / 256, 256, 0, stream>>>(Ws[i], wb[i], DMODEL * DMODEL);

  const float* bias3[3]  = {bq, bk, bv};
  const float* gam3[3]   = {gq, gk, gv};
  const float* beta3[3]  = {betaq, betak, betav};
  uint8_t* dst3[3]       = {qbin, kbin, vTb};
  for (int t = 0; t < 3; ++t) {
    gemm_bf16_kernel<<<gemmGrid, 256, 0, stream>>>(xb, wb[t], bias3[t], Y);
    bn_stats_kernel<<<DMODEL / 256, 256, 0, stream>>>(Y, gam3[t], beta3[t], colA, colB);
    binarize_kernel<<<(ROWS * DMODEL) / 256, 256, 0, stream>>>(
        Y, colA, colB, dst3[t], nullptr, (t == 2) ? 1 : 0);
  }

  energy_kernel<<<NBATCH * NHEAD, 256, 0, stream>>>(qbin, kbin, ener);
  gate_kernel<<<1, 256, 0, stream>>>(Wg, bg, ener, gat);

  attention_kernel<<<dim3(NBATCH * NHEAD, N_TOK / 128), 256, 0, stream>>>(
      qbin, kbin, vTb, gat, s2);

  gemm_bf16_kernel<<<gemmGrid, 256, 0, stream>>>(s2, wb[3], bp, Y);
  bn_stats_kernel<<<DMODEL / 256, 256, 0, stream>>>(Y, gp, betap, colA, colB);
  binarize_kernel<<<(ROWS * DMODEL) / 256, 256, 0, stream>>>(
      Y, colA, colB, nullptr, (float*)d_out, 2);

  (void)in_sizes; (void)n_in; (void)out_size; (void)ws_size;
}
